// GraphEncoderBlock_81896436400166
// MI455X (gfx1250) — matrix operand order repaired and checked by reading, not compiled, not run on hardware
//
#include <hip/hip_runtime.h>
#include <hip/hip_bf16.h>
#include <math.h>

#define NN 50000
#define EE 400000
#define DD 128
#define HH 4
#define EDIM 16

typedef __attribute__((ext_vector_type(16))) _Float16 v16h;
typedef __attribute__((ext_vector_type(8)))  _Float16 v8h;
typedef __attribute__((ext_vector_type(8)))  float    v8f;
typedef __attribute__((ext_vector_type(4)))  unsigned v4u;
typedef __attribute__((ext_vector_type(8)))  int      v8i;
typedef __attribute__((ext_vector_type(4)))  int      v4i;

// ---------- helpers ----------

// Monotonic float->uint mapping for atomic max on signed floats
__device__ __forceinline__ unsigned ordmap(float f) {
    unsigned u = __float_as_uint(f);
    unsigned mask = (unsigned)((int)u >> 31) | 0x80000000u;
    return u ^ mask;
}
__device__ __forceinline__ float orddecode(unsigned u) {
    unsigned mask = (u & 0x80000000u) ? 0x80000000u : 0xFFFFFFFFu;
    return __uint_as_float(u ^ mask);
}

// A-matrix (16x32 f16, M=lane&15) fragment from an f32 row pointer.
// element e: K = 32*kc + 16*lh + e
__device__ __forceinline__ v16h load_a_f32(const float* rowp, int kc, int lh) {
    const float4 q0 = *(const float4*)(rowp + 32 * kc + 16 * lh);
    const float4 q1 = *(const float4*)(rowp + 32 * kc + 16 * lh + 4);
    const float4 q2 = *(const float4*)(rowp + 32 * kc + 16 * lh + 8);
    const float4 q3 = *(const float4*)(rowp + 32 * kc + 16 * lh + 12);
    v16h a;
    a[0] = (_Float16)q0.x; a[1] = (_Float16)q0.y; a[2] = (_Float16)q0.z; a[3] = (_Float16)q0.w;
    a[4] = (_Float16)q1.x; a[5] = (_Float16)q1.y; a[6] = (_Float16)q1.z; a[7] = (_Float16)q1.w;
    a[8]  = (_Float16)q2.x; a[9]  = (_Float16)q2.y; a[10] = (_Float16)q2.z; a[11] = (_Float16)q2.w;
    a[12] = (_Float16)q3.x; a[13] = (_Float16)q3.y; a[14] = (_Float16)q3.z; a[15] = (_Float16)q3.w;
    return a;
}

// Same fragment pattern but from an f16 row pointer (LDS).
__device__ __forceinline__ v16h load_a_h(const _Float16* rowp, int kc, int lh) {
    v8h lo = *(const v8h*)(rowp + 32 * kc + 16 * lh);
    v8h hi = *(const v8h*)(rowp + 32 * kc + 16 * lh + 8);
    v16h a;
#pragma unroll
    for (int i = 0; i < 8; i++) { a[i] = lo[i]; a[8 + i] = hi[i]; }
    return a;
}

// ---------- K0: init scratch + convert weights to f16 ----------
__global__ __launch_bounds__(256) void k_init(
    float* __restrict__ hbuf, float* __restrict__ den, unsigned* __restrict__ mbuf,
    const float* __restrict__ Wl, const float* __restrict__ Wr,
    const float* __restrict__ W1, const float* __restrict__ W2,
    _Float16* __restrict__ Wl_h, _Float16* __restrict__ Wr_h,
    _Float16* __restrict__ W1_h, _Float16* __restrict__ W2_h) {
    int i = blockIdx.x * 256 + threadIdx.x;
    if (i < NN * DD) hbuf[i] = 0.f;
    if (i < NN * HH) { den[i] = 0.f; mbuf[i] = 0x007FFFFFu; /* enc(-inf) */ }
    if (i < 512 * 128) { Wl_h[i] = (_Float16)Wl[i]; Wr_h[i] = (_Float16)Wr[i]; }
    if (i < 256 * 128) { W1_h[i] = (_Float16)W1[i]; W2_h[i] = (_Float16)W2[i]; }
}

// ---------- K1: xl = x@Wl.T + bl ; xr = x@Wr.T + br  (WMMA) ----------
__global__ __launch_bounds__(32) void k_node_gemm(
    const float* __restrict__ x,
    const _Float16* __restrict__ Wl_h, const float* __restrict__ bl,
    const _Float16* __restrict__ Wr_h, const float* __restrict__ br,
    float* __restrict__ xl, float* __restrict__ xr) {
    const int lane = threadIdx.x, lm = lane & 15, lh = lane >> 4;
    const int node0 = blockIdx.x * 16;
    const float* rowp = x + (size_t)(node0 + lm) * DD;
    v16h a[4];
#pragma unroll
    for (int kc = 0; kc < 4; kc++) a[kc] = load_a_f32(rowp, kc, lh);
    for (int sel = 0; sel < 2; sel++) {
        const _Float16* W = sel ? Wr_h : Wl_h;
        const float* bias = sel ? br : bl;
        float* out = sel ? xr : xl;
        for (int t = 0; t < 32; t++) {
            v8f c = {};
#pragma unroll
            for (int kc = 0; kc < 4; kc++) {
                v16h b = *(const v16h*)(W + (size_t)(t * 16 + lm) * 128 + 32 * kc + 16 * lh);
                c = __builtin_amdgcn_wmma_f32_16x16x32_f16(false, a[kc], false, b,
                                                           (short)0, c, false, false);
            }
            float bv = bias[t * 16 + lm];
#pragma unroll
            for (int r = 0; r < 8; r++)
                out[(size_t)(node0 + r + 8 * lh) * 512 + t * 16 + lm] = c[r] + bv;
        }
    }
}

// ---------- K2: per-edge attention logits + segment max ----------
// We (512x16 f32, 32 KB) is staged into LDS by the Tensor Data Mover with
// pad_enable: 1 dword pad per 16-dword row -> LDS pitch 17 dwords, which makes
// the per-lane reads We_lds[d*17+k] bank-conflict-free (gcd(17,64)=1).
__global__ __launch_bounds__(256) void k_edge_logits(
    const int* __restrict__ ei, const float* __restrict__ ea,
    const float* __restrict__ xl, const float* __restrict__ xr,
    const float* __restrict__ We, const float* __restrict__ att,
    float* __restrict__ logit, unsigned* __restrict__ mbuf) {
    __shared__ float We_lds[512 * 17];  // padded pitch-17 rows, ~34 KB
    __shared__ float att_s[512];
    const int tid = threadIdx.x;
    const int wave = tid >> 5, lane = tid & 31;
    const int e = blockIdx.x * 8 + wave;
    const int src = ei[e], dst = ei[EE + e];
    const float* xls = xl + (size_t)src * 512;
    const float* xrd = xr + (size_t)dst * 512;

    // Prefetch the gathered rows (lane-strided: 32 lanes cover the 2 KB row)
    __builtin_prefetch(xls + lane * 16, 0, 0);
    __builtin_prefetch(xrd + lane * 16, 0, 0);

    if (wave == 0) {
        // TDM descriptor: 2D tile 16 x 512 dwords, data_size=4B,
        // pad_enable=1, pad_interval=3 (every 16 dwords), pad_amount=0 (+1 dword)
        unsigned long long ga = (unsigned long long)(uintptr_t)We;
        unsigned lds_addr = (unsigned)(uintptr_t)We_lds;
        v4u g0;
        g0[0] = 1u;                                   // count=1 (valid), user mode
        g0[1] = lds_addr;                             // lds_addr
        g0[2] = (unsigned)ga;                         // global_addr[31:0]
        g0[3] = (unsigned)(ga >> 32) | (2u << 30);    // global_addr[56:32] | type=2
        v8i g1;
        g1[0] = (int)((2u << 16) | (1u << 20) | (3u << 22)); // data_size=4B, pad_en, interval=16dw
        g1[1] = (int)(16u << 16);   // tensor_dim0 = 16 (low bits -> [63:48])
        g1[2] = (int)(512u << 16);  // tensor_dim1 = 512
        g1[3] = (int)(16u << 16);   // tile_dim0 = 16  ([127:112])
        g1[4] = (int)512;           // tile_dim1 = 512 ([143:128]); tile_dim2 = 0
        g1[5] = (int)16;            // tensor_dim0_stride = 16
        g1[6] = 0;
        g1[7] = 0;
        v4i g2 = {0, 0, 0, 0};      // higher dims unused
        v4i g3 = {0, 0, 0, 0};
        v8i g4 = {0, 0, 0, 0, 0, 0, 0, 0};  // extra group (6-arg toolchain form)
        __builtin_amdgcn_tensor_load_to_lds(g0, g1, g2, g3, g4, 0);
        __builtin_amdgcn_s_wait_tensorcnt(0);
    }
    for (int i = tid; i < 512; i += 256) att_s[i] = att[i];
    __syncthreads();

    float eav[EDIM];
#pragma unroll
    for (int k = 0; k < EDIM; k++) eav[k] = ea[(size_t)e * EDIM + k];
    float acc[HH] = {0.f, 0.f, 0.f, 0.f};
#pragma unroll
    for (int j = 0; j < 16; j++) {
        const int d = 32 * j + lane;
        float ev = 0.f;
#pragma unroll
        for (int k = 0; k < EDIM; k++) ev = fmaf(eav[k], We_lds[d * 17 + k], ev);
        float z = xls[d] + xrd[d] + ev;
        z = z > 0.f ? z : 0.2f * z;
        acc[j >> 2] = fmaf(att_s[d], z, acc[j >> 2]);   // att flat[h*128+d%128] == att[d]
    }
#pragma unroll
    for (int off = 16; off > 0; off >>= 1) {
#pragma unroll
        for (int h = 0; h < HH; h++) acc[h] += __shfl_xor(acc[h], off, 32);
    }
    if (lane == 0) {
#pragma unroll
        for (int h = 0; h < HH; h++) {
            logit[(size_t)e * HH + h] = acc[h];
            atomicMax(&mbuf[dst * HH + h], ordmap(acc[h]));
        }
    }
}

// ---------- K3: ex = exp(logit - m[dst]); den[dst] += ex ----------
__global__ __launch_bounds__(256) void k_softmax_den(
    const int* __restrict__ ei, const unsigned* __restrict__ mbuf,
    float* __restrict__ logit, float* __restrict__ den) {
    int t = blockIdx.x * 256 + threadIdx.x;     // t < E*H
    int e = t >> 2, h = t & 3;
    int dst = ei[EE + e];
    float m = orddecode(mbuf[dst * HH + h]);
    float ex = expf(logit[t] - m);
    logit[t] = ex;                              // reuse buffer for ex
    atomicAdd(&den[dst * HH + h], ex);
}

// ---------- K4: weighted scatter, head-folded mean ----------
__global__ __launch_bounds__(256) void k_scatter(
    const int* __restrict__ ei, const float* __restrict__ exb,
    const float* __restrict__ den, const float* __restrict__ xl,
    float* __restrict__ hbuf) {
    const int wave = threadIdx.x >> 5, lane = threadIdx.x & 31;
    const int e = blockIdx.x * 8 + wave;
    const int src = ei[e], dst = ei[EE + e];
    float a[HH];
#pragma unroll
    for (int h = 0; h < HH; h++)
        a[h] = exb[(size_t)e * HH + h] / (den[dst * HH + h] + 1e-16f) * 0.25f; // *1/H
    const float* xls = xl + (size_t)src * 512;
#pragma unroll
    for (int dd = 0; dd < 4; dd++) {
        const int d2 = 32 * dd + lane;
        float s = 0.f;
#pragma unroll
        for (int h = 0; h < HH; h++) s = fmaf(xls[h * 128 + d2], a[h], s);
        atomicAdd(&hbuf[(size_t)dst * DD + d2], s);
    }
}

// ---------- K5: x1 = LN(x + h + conv_bias) ----------
__global__ __launch_bounds__(128) void k_ln1(
    const float* __restrict__ x, const float* __restrict__ hbuf,
    const float* __restrict__ cb, const float* __restrict__ g1,
    const float* __restrict__ be1, float* __restrict__ x1) {
    __shared__ float r1[128], r2[128];
    const int n = blockIdx.x, d = threadIdx.x;
    float v = x[(size_t)n * DD + d] + hbuf[(size_t)n * DD + d] + cb[d];
    r1[d] = v; r2[d] = v * v;
    __syncthreads();
    for (int s = 64; s > 0; s >>= 1) {
        if (d < s) { r1[d] += r1[d + s]; r2[d] += r2[d + s]; }
        __syncthreads();
    }
    float mu = r1[0] * (1.f / 128.f);
    float var = r2[0] * (1.f / 128.f) - mu * mu;
    x1[(size_t)n * DD + d] = (v - mu) * rsqrtf(var + 1e-5f) * g1[d] + be1[d];
}

// ---------- K6: fused FFN (WMMA) + residual + LN2 ----------
__global__ __launch_bounds__(32) void k_ffn(
    const float* __restrict__ x1,
    const _Float16* __restrict__ W1_h, const float* __restrict__ b1,
    const _Float16* __restrict__ W2_h, const float* __restrict__ b2,
    const float* __restrict__ g2, const float* __restrict__ be2,
    float* __restrict__ out) {
    __shared__ _Float16 y_lds[16 * 256];   // GELU activations, f16
    __shared__ float    o_lds[16 * 128];   // GEMM2 output, f32
    const int lane = threadIdx.x, lm = lane & 15, lh = lane >> 4;
    const int node0 = blockIdx.x * 16;
    const float* rowp = x1 + (size_t)(node0 + lm) * DD;
    v16h a1[4];
#pragma unroll
    for (int kc = 0; kc < 4; kc++) a1[kc] = load_a_f32(rowp, kc, lh);

    // GEMM1: 16x256 = x1_tile @ W1.T, + b1, exact GELU
    for (int t = 0; t < 16; t++) {
        v8f c = {};
#pragma unroll
        for (int kc = 0; kc < 4; kc++) {
            v16h b = *(const v16h*)(W1_h + (size_t)(t * 16 + lm) * 128 + 32 * kc + 16 * lh);
            c = __builtin_amdgcn_wmma_f32_16x16x32_f16(false, a1[kc], false, b,
                                                       (short)0, c, false, false);
        }
        float bv = b1[t * 16 + lm];
#pragma unroll
        for (int r = 0; r < 8; r++) {
            float v = c[r] + bv;
            float g = 0.5f * v * (1.f + erff(v * 0.70710678118f));
            y_lds[(size_t)(r + 8 * lh) * 256 + t * 16 + lm] = (_Float16)g;
        }
    }
    __syncthreads();

    // GEMM2: 16x128 = y @ W2.T, + b2
    v16h a2[8];
    const _Float16* yrow = y_lds + (size_t)lm * 256;
#pragma unroll
    for (int kc = 0; kc < 8; kc++) a2[kc] = load_a_h(yrow, kc, lh);
    for (int t2 = 0; t2 < 8; t2++) {
        v8f c = {};
#pragma unroll
        for (int kc = 0; kc < 8; kc++) {
            v16h b = *(const v16h*)(W2_h + (size_t)(t2 * 16 + lm) * 256 + 32 * kc + 16 * lh);
            c = __builtin_amdgcn_wmma_f32_16x16x32_f16(false, a2[kc], false, b,
                                                       (short)0, c, false, false);
        }
        float bv = b2[t2 * 16 + lm];
#pragma unroll
        for (int r = 0; r < 8; r++)
            o_lds[(size_t)(r + 8 * lh) * 128 + t2 * 16 + lm] = c[r] + bv;
    }
    __syncthreads();

    // residual + LayerNorm2, one row at a time (wave-wide shuffle reduce)
    for (int m = 0; m < 16; m++) {
        float v[4], s = 0.f, s2 = 0.f;
#pragma unroll
        for (int q = 0; q < 4; q++) {
            const int d2 = lane + 32 * q;
            v[q] = o_lds[m * 128 + d2] + x1[(size_t)(node0 + m) * DD + d2];
            s += v[q]; s2 += v[q] * v[q];
        }
#pragma unroll
        for (int off = 16; off > 0; off >>= 1) {
            s += __shfl_xor(s, off, 32);
            s2 += __shfl_xor(s2, off, 32);
        }
        float mu = s * (1.f / 128.f);
        float var = s2 * (1.f / 128.f) - mu * mu;
        float rs = rsqrtf(var + 1e-5f);
#pragma unroll
        for (int q = 0; q < 4; q++) {
            const int d2 = lane + 32 * q;
            out[(size_t)(node0 + m) * DD + d2] = (v[q] - mu) * rs * g2[d2] + be2[d2];
        }
    }
}

// ---------- launch ----------
extern "C" void kernel_launch(void* const* d_in, const int* in_sizes, int n_in,
                              void* d_out, int out_size, void* d_ws, size_t ws_size,
                              hipStream_t stream) {
    const float* x   = (const float*)d_in[0];
    const int*   ei  = (const int*)d_in[1];
    const float* ea  = (const float*)d_in[2];
    const float* Wl  = (const float*)d_in[3];
    const float* bl  = (const float*)d_in[4];
    const float* Wr  = (const float*)d_in[5];
    const float* br  = (const float*)d_in[6];
    const float* We  = (const float*)d_in[7];
    const float* att = (const float*)d_in[8];
    const float* cb  = (const float*)d_in[9];
    const float* g1  = (const float*)d_in[10];
    const float* be1 = (const float*)d_in[11];
    const float* W1  = (const float*)d_in[12];
    const float* b1  = (const float*)d_in[13];
    const float* W2  = (const float*)d_in[14];
    const float* b2  = (const float*)d_in[15];
    const float* g2  = (const float*)d_in[16];
    const float* be2 = (const float*)d_in[17];
    float* out = (float*)d_out;

    char* w = (char*)d_ws;
    auto take = [&](size_t bytes) -> char* {
        char* p = w;
        w += (bytes + 255) & ~(size_t)255;
        return p;
    };
    float*     xl    = (float*)take((size_t)NN * 512 * 4);
    float*     xr    = (float*)take((size_t)NN * 512 * 4);
    float*     logit = (float*)take((size_t)EE * HH * 4);
    unsigned*  mbuf  = (unsigned*)take((size_t)NN * HH * 4);
    float*     den   = (float*)take((size_t)NN * HH * 4);
    float*     hbuf  = (float*)take((size_t)NN * DD * 4);
    float*     x1    = (float*)take((size_t)NN * DD * 4);
    _Float16*  Wl_h  = (_Float16*)take(512 * 128 * 2);
    _Float16*  Wr_h  = (_Float16*)take(512 * 128 * 2);
    _Float16*  W1_h  = (_Float16*)take(256 * 128 * 2);
    _Float16*  W2_h  = (_Float16*)take(128 * 256 * 2);

    k_init<<<(NN * DD + 255) / 256, 256, 0, stream>>>(hbuf, den, mbuf, Wl, Wr, W1, W2,
                                                      Wl_h, Wr_h, W1_h, W2_h);
    k_node_gemm<<<NN / 16, 32, 0, stream>>>(x, Wl_h, bl, Wr_h, br, xl, xr);
    k_edge_logits<<<EE / 8, 256, 0, stream>>>(ei, ea, xl, xr, We, att, logit, mbuf);
    k_softmax_den<<<(EE * HH) / 256, 256, 0, stream>>>(ei, mbuf, logit, den);
    k_scatter<<<EE / 8, 256, 0, stream>>>(ei, logit, den, xl, hbuf);
    k_ln1<<<NN, 128, 0, stream>>>(x, hbuf, cb, g1, be1, x1);
    k_ffn<<<NN / 16, 32, 0, stream>>>(x1, W1_h, b1, W2_h, b2, g2, be2, out);
}
